// ExternalLinear_30571577213536
// MI455X (gfx1250) — matrix operand order repaired and checked by reading, not compiled, not run on hardware
//
#include <hip/hip_runtime.h>

typedef __attribute__((ext_vector_type(16))) __bf16 v16bf;
typedef __attribute__((ext_vector_type(8)))  float  v8f;

#define WAVES            8
#define STRIPS_PER_WAVE  4
#define LDS_STRIDE       136   // 128 bf16 + 8 pad -> 68 dwords/row: conflict-free b128 reads

union Frag {
    v16bf    v;
    unsigned u[8];
    uint4    q[2];
};

// Two fp32 -> packed bf16 pair (lo in [15:0], hi in [31:16]).
// Round-half-up (+0x8000) then one v_perm_b32 grabs the high halves of both:
// result bytes {3,2} = uhi bytes {3,2}, bytes {1,0} = ulo bytes {3,2}.
__device__ __forceinline__ unsigned bf16pack(float lo, float hi) {
    const unsigned ulo = __float_as_uint(lo) + 0x8000u;
    const unsigned uhi = __float_as_uint(hi) + 0x8000u;
    return __builtin_amdgcn_perm(uhi, ulo, 0x07060302u);
}

extern "C" __global__ __launch_bounds__(256)
void linear_wmma_bf16(const float* __restrict__ x,
                      const float* __restrict__ w,
                      const float* __restrict__ bias,
                      float* __restrict__ out,
                      int nrows)
{
    __shared__ unsigned short ldsW[128 * LDS_STRIDE];

    const int tid = threadIdx.x;

    // ---- Stage W (fp32 row-major [N=128][K=128]) into LDS as bf16, padded rows.
    // 128*128 elems / 4-per-iter / 256 threads = 16 iterations, float4 loads.
    for (int idx = tid; idx < 128 * 32; idx += 256) {
        const int n  = idx >> 5;
        const int k4 = (idx & 31) << 2;
        const float4 f = *(const float4*)(w + n * 128 + k4);
        const unsigned p0 = bf16pack(f.x, f.y);
        const unsigned p1 = bf16pack(f.z, f.w);
        *(uint2*)&ldsW[n * LDS_STRIDE + k4] = make_uint2(p0, p1);
    }
    __syncthreads();

    const int lane = tid & 31;
    const int wave = tid >> 5;
    const int col  = lane & 15;   // N index (B/C/D), M row index (A)
    const int hl   = lane >> 4;   // lane-half

    // Bias per lane, per N-tile: D layout puts N = lane&15 in every acc VGPR.
    float bv[8];
    #pragma unroll
    for (int nt = 0; nt < 8; ++nt) bv[nt] = bias[nt * 16 + col];

    for (int it = 0; it < STRIPS_PER_WAVE; ++it) {
        const int strip = (blockIdx.x * STRIPS_PER_WAVE + it) * WAVES + wave;
        const int m0 = strip * 16;
        if (m0 >= nrows) break;                     // uniform per wave

        // ---- A fragments: x rows m0..m0+15, converted fp32->bf16 in-flight.
        // ISA 16-bit A 16x32 layout: lane-half 0 holds K {k0..k0+7, k0+16..k0+23},
        // half 1 holds {k0+8..k0+15, k0+24..k0+31}.
        const float* xr = x + (size_t)(m0 + col) * 128;
        Frag a[4];
        #pragma unroll
        for (int kk = 0; kk < 4; ++kk) {
            const int b0 = kk * 32 + hl * 8;
            const float4 f0 = *(const float4*)(xr + b0);
            const float4 f1 = *(const float4*)(xr + b0 + 4);
            const float4 f2 = *(const float4*)(xr + b0 + 16);
            const float4 f3 = *(const float4*)(xr + b0 + 20);
            a[kk].u[0] = bf16pack(f0.x, f0.y);
            a[kk].u[1] = bf16pack(f0.z, f0.w);
            a[kk].u[2] = bf16pack(f1.x, f1.y);
            a[kk].u[3] = bf16pack(f1.z, f1.w);
            a[kk].u[4] = bf16pack(f2.x, f2.y);
            a[kk].u[5] = bf16pack(f2.z, f2.w);
            a[kk].u[6] = bf16pack(f3.x, f3.y);
            a[kk].u[7] = bf16pack(f3.z, f3.w);
        }

        // ---- 8 N-tiles x 4 K-steps of v_wmma_f32_16x16x32_bf16
        #pragma unroll
        for (int nt = 0; nt < 8; ++nt) {
            const int n0 = nt * 16;
            v8f acc = {bv[nt], bv[nt], bv[nt], bv[nt],
                       bv[nt], bv[nt], bv[nt], bv[nt]};
            #pragma unroll
            for (int kk = 0; kk < 4; ++kk) {
                // B layout: lane-half h holds K {k0+8h..k0+8h+7, k0+16+8h..k0+16+8h+7}
                // for column N = lane&15; B[k,n] = W[n,k] -> two 16B chunks in LDS.
                Frag b;
                const uint4* p = (const uint4*)
                    &ldsW[(n0 + col) * LDS_STRIDE + kk * 32 + hl * 8];
                b.q[0] = p[0];
                b.q[1] = p[2];
                acc = __builtin_amdgcn_wmma_f32_16x16x32_bf16(
                        false, a[kk].v, false, b.v, (short)0, acc, false, false);
            }
            // D layout: VGPR v -> row m0 + v + 8*hl, col n0 + (lane&15)
            float* orow = out + (size_t)(m0 + hl * 8) * 128 + n0 + col;
            #pragma unroll
            for (int v = 0; v < 8; ++v) orow[(size_t)v * 128] = acc[v];
        }
    }
}

extern "C" void kernel_launch(void* const* d_in, const int* in_sizes, int n_in,
                              void* d_out, int out_size, void* d_ws, size_t ws_size,
                              hipStream_t stream) {
    const float* x    = (const float*)d_in[0];
    const float* w    = (const float*)d_in[1];
    const float* bias = (const float*)d_in[2];
    float* out        = (float*)d_out;

    const int nrows  = in_sizes[0] / 128;                 // 262144
    const int strips = (nrows + 15) / 16;                 // 16384
    const int per_blk = WAVES * STRIPS_PER_WAVE;          // 32
    const int blocks  = (strips + per_blk - 1) / per_blk; // 512

    hipLaunchKernelGGL(linear_wmma_bf16, dim3(blocks), dim3(256), 0, stream,
                       x, w, bias, out, nrows);
}
